// Transformer_52175262712059
// MI455X (gfx1250) — matrix operand order repaired and checked by reading, not compiled, not run on hardware
//
#include <hip/hip_runtime.h>

typedef unsigned short u16;
typedef unsigned int   u32;
typedef unsigned long long u64;

typedef __attribute__((ext_vector_type(16))) __bf16 v16bf;
typedef __attribute__((ext_vector_type(2)))  __bf16 v2bf;
typedef __attribute__((ext_vector_type(8)))  float  v8f;
typedef __attribute__((ext_vector_type(4)))  u32    v4u;
typedef __attribute__((ext_vector_type(8)))  u32    v8u;

union Frag {
    v16bf v;
    uint4 u[2];
    u16   h[16];
};

static __device__ __forceinline__ u16 f2bf(float f) {
    u32 u = __float_as_uint(f);
    u32 r = (u + 0x7FFFu + ((u >> 16) & 1u)) >> 16;  // RNE
    return (u16)r;
}

static __device__ __forceinline__ u32 packbf(float lo, float hi) {
#if __has_builtin(__builtin_amdgcn_cvt_pk_bf16_f32)
    union { v2bf v; u32 u; } p;
    p.v = __builtin_amdgcn_cvt_pk_bf16_f32(lo, hi);
    return p.u;
#else
    return (u32)f2bf(lo) | ((u32)f2bf(hi) << 16);
#endif
}

// ---------------------------------------------------------------------------
// TDM: issue a 2D bf16 tensor_load_to_lds (tile 32 x rows, padded LDS rows).
// D# per CDNA5 ISA 8.3/8.4. data_size=2B; pad_enable: pad_interval=16 DWORDs
// (one 64B tile row), pad_amount=4 DWORDs (16B) -> LDS row stride 80B (LDP=40).
// ---------------------------------------------------------------------------
#define LDP 40

static __device__ __forceinline__ void tdm_load_tile_a(
    u32 lds_off, const void* gaddr, u32 td0 /*tensor K*/, u32 td1 /*tensor M*/,
    u32 stride /*elements*/) {
    u64 ga = (u64)(uintptr_t)gaddr;
    v4u g0;
    g0.x = 1u;                               // count=1, user mode, no gather
    g0.y = lds_off;                          // lds_addr (bytes)
    g0.z = (u32)ga;                          // global_addr[31:0]
    g0.w = (u32)(ga >> 32) | 0x80000000u;    // global_addr[56:32] | type=2
    v8u g1;
    g1[0] = (1u << 16)      // data_size = 2 bytes
          | (1u << 20)      // pad_enable
          | (3u << 22)      // pad_interval: 16 DWORDs
          | (3u << 25);     // pad_amount:   4 DWORDs
    g1[1] = (td0 & 0xFFFFu) << 16;                  // tensor_dim0 lo16
    g1[2] = (td0 >> 16) | ((td1 & 0xFFFFu) << 16);  // td0 hi16 | tensor_dim1 lo16
    g1[3] = (td1 >> 16) | (32u << 16);              // td1 hi16 | tile_dim0 = 32
    g1[4] = 128u;                                   // tile_dim1 = 128 (tile_dim2=0)
    g1[5] = stride;                                 // tensor_dim0_stride lo32
    g1[6] = 0u;                                     // stride hi16 | dim1_stride lo16
    g1[7] = 0u;                                     // dim1_stride hi32
    asm volatile("tensor_load_to_lds %0, %1" :: "s"(g0), "s"(g1) : "memory");
}

// ---------------------------------------------------------------------------
// LayerNorm (fp32 in) -> bf16 out.  One block per token row, DIM=1024.
// ---------------------------------------------------------------------------
__global__ __launch_bounds__(256)
void ln_bf16(const float* __restrict__ h, const float* __restrict__ g,
             const float* __restrict__ bb, u16* __restrict__ y) {
    __shared__ float red[16];
    const int row = blockIdx.x;
    const int t = threadIdx.x;
    const float* hp = h + (size_t)row * 1024;
    float4 v = *(const float4*)&hp[t * 4];
    float s  = v.x + v.y + v.z + v.w;
    float ss = v.x * v.x + v.y * v.y + v.z * v.z + v.w * v.w;
    for (int off = 16; off; off >>= 1) {
        s  += __shfl_down(s,  off, 32);
        ss += __shfl_down(ss, off, 32);
    }
    const int wave = t >> 5, lane = t & 31;
    if (lane == 0) { red[wave] = s; red[wave + 8] = ss; }
    __syncthreads();
    if (t == 0) {
        float S = 0.f, SS = 0.f;
        for (int i = 0; i < 8; ++i) { S += red[i]; SS += red[i + 8]; }
        red[0] = S * (1.0f / 1024.0f);
        red[1] = SS * (1.0f / 1024.0f);
    }
    __syncthreads();
    const float mu = red[0];
    const float rstd = rsqrtf(red[1] - mu * mu + 1e-3f);
    u16* yp = y + (size_t)row * 1024;
    const int c = t * 4;
    u32* yp32 = (u32*)(yp + c);
    yp32[0] = packbf((v.x - mu) * rstd * g[c + 0] + bb[c + 0],
                     (v.y - mu) * rstd * g[c + 1] + bb[c + 1]);
    yp32[1] = packbf((v.z - mu) * rstd * g[c + 2] + bb[c + 2],
                     (v.w - mu) * rstd * g[c + 3] + bb[c + 3]);
}

// ---------------------------------------------------------------------------
// GEMM: C[M,Nw] = epi(A_bf16[M,K] @ W_f32[K,Nw] + bias)
// MODE 0: store bf16   MODE 1: gelu -> bf16   MODE 2: fp32 residual add
// Block 256 thr = 8 waves (4x2), tile 128x128, K-step 32.
// A tile staged by TDM (double buffered, overlaps compute); W tile converted
// fp32->bf16 and transposed into LDS with packed b32 stores (double buffered).
// ---------------------------------------------------------------------------
template <int MODE>
__global__ __launch_bounds__(256)
void gemm_bf16(const u16* __restrict__ A, const float* __restrict__ W,
               const float* __restrict__ bias, void* __restrict__ outp,
               int K, int Nw) {
    __shared__ u16 Asl[2][128 * LDP];
    __shared__ u16 Bsl[2][128 * LDP];
    const int t    = threadIdx.x;
    const int lane = t & 31;
    const int wave = t >> 5;
    const int wm   = wave >> 1;   // 0..3
    const int wn   = wave & 1;    // 0..1
    const int hl   = lane >> 4;
    const int lr   = lane & 15;
    const int m0   = blockIdx.y * 128;
    const int n0   = blockIdx.x * 128;

    const u32 aoff[2] = { (u32)(uintptr_t)&Asl[0][0], (u32)(uintptr_t)&Asl[1][0] };

    v8f acc[2][4];
    const v8f vzero = {0.f, 0.f, 0.f, 0.f, 0.f, 0.f, 0.f, 0.f};
    for (int i = 0; i < 2; ++i)
        for (int j = 0; j < 4; ++j) acc[i][j] = vzero;

    const int nIter = K >> 5;
    if (wave == 0)
        tdm_load_tile_a(aoff[0], A + (size_t)m0 * K, (u32)K, 4096u, (u32)K);

    for (int i = 0; i < nIter; ++i) {
        const int cur = i & 1;
        const int k0  = i << 5;
        // ---- stage W tile 32x128 fp32 -> Bsl[cur][n][k] bf16 (packed pairs)
        u32* bs = (u32*)&Bsl[cur][0];
        for (int q = 0; q < 2; ++q) {
            int idx = t + q * 256;            // 0..511 cells (4 n  x 2 k each)
            int n4 = idx & 31, kp = idx >> 5; // kp: 0..15
            const float* wp0 = &W[(size_t)(k0 + 2 * kp) * Nw + n0 + n4 * 4];
            float4 w0 = *(const float4*)wp0;
            float4 w1 = *(const float4*)(wp0 + Nw);
            if (k0 + 32 < K)
                __builtin_prefetch((const void*)(wp0 + (size_t)32 * Nw), 0, 1);
            bs[(n4 * 4 + 0) * (LDP / 2) + kp] = packbf(w0.x, w1.x);
            bs[(n4 * 4 + 1) * (LDP / 2) + kp] = packbf(w0.y, w1.y);
            bs[(n4 * 4 + 2) * (LDP / 2) + kp] = packbf(w0.z, w1.z);
            bs[(n4 * 4 + 3) * (LDP / 2) + kp] = packbf(w0.w, w1.w);
        }
        if (wave == 0) __builtin_amdgcn_s_wait_tensorcnt(0);  // A[cur] landed
        __syncthreads();  // everyone sees A[cur] + B[cur]; all done with buf^1
        if (wave == 0 && i + 1 < nIter)   // prefetch next A tile via TDM
            tdm_load_tile_a(aoff[cur ^ 1], A + (size_t)m0 * K + (k0 + 32),
                            (u32)K, 4096u, (u32)K);

        // ---- compute
        Frag bf[4];
        for (int ni = 0; ni < 4; ++ni) {
            int c = wn * 64 + ni * 16 + lr;
            int base = c * LDP + hl * 16;   // lanes<16: K0..15, >=16: K16..31
            bf[ni].u[0] = *(const uint4*)&Bsl[cur][base];
            bf[ni].u[1] = *(const uint4*)&Bsl[cur][base + 8];
        }
        for (int mi = 0; mi < 2; ++mi) {
            Frag af;
            int r = wm * 32 + mi * 16 + lr;
            int base = r * LDP + hl * 16;   // lanes<16: K0..15, >=16: K16..31
            af.u[0] = *(const uint4*)&Asl[cur][base];
            af.u[1] = *(const uint4*)&Asl[cur][base + 8];
            for (int ni = 0; ni < 4; ++ni)
                acc[mi][ni] = __builtin_amdgcn_wmma_f32_16x16x32_bf16(
                    false, af.v, false, bf[ni].v, (short)0, acc[mi][ni], false, false);
        }
    }

    for (int mi = 0; mi < 2; ++mi)
        for (int ni = 0; ni < 4; ++ni) {
            int gc = n0 + wn * 64 + ni * 16 + lr;
            float bv = bias[gc];
            int rb = m0 + wm * 32 + mi * 16 + hl * 8;
            if (MODE == 2) {
                float* of = (float*)outp;
                for (int r = 0; r < 8; ++r)
                    of[(size_t)(rb + r) * Nw + gc] += acc[mi][ni][r] + bv;
            } else {
                u16* ob = (u16*)outp;
                for (int r = 0; r < 8; ++r) {
                    float vv = acc[mi][ni][r] + bv;
                    if (MODE == 1) vv = 0.5f * vv * (1.0f + erff(vv * 0.70710678f));
                    ob[(size_t)(rb + r) * Nw + gc] = f2bf(vv);
                }
            }
        }
}

// ---------------------------------------------------------------------------
// Flash attention. qkv bf16 [B,N,3072] (q|k|v each [H=16][DH=64]).
// Block = 4 waves; wave = 16 queries x full head. S^T = K*Q^T (softmax rows
// in lanes, one shfl_xor(16) reduce), then O^T = V^T * P^T with in-register
// P^T B-fragment. Grid: (N/64, HEADS, B)
// ---------------------------------------------------------------------------
__global__ __launch_bounds__(128)
void attn(const u16* __restrict__ qkv, u16* __restrict__ o) {
    __shared__ u16 Vt[64 * LDP];  // V^T [dh][key] for a 32-key tile
    const int t = threadIdx.x, lane = t & 31, wave = t >> 5;
    const int hl = lane >> 4, lr = lane & 15;
    const int chunk = blockIdx.x, head = blockIdx.y, b = blockIdx.z;
    const int qbase = chunk * 64 + wave * 16;
    const size_t RS = 3072;
    const float SC = 1.0f / (1024.0f * 1024.0f);  // 1/(DIM*DIM), per reference

    const u16* qp = qkv + ((size_t)(b * 2048 + qbase + lr)) * RS + head * 64;
    Frag qb[2];
    for (int s = 0; s < 2; ++s) {
        qb[s].u[0] = *(const uint4*)(qp + s * 32 + hl * 16);
        qb[s].u[1] = *(const uint4*)(qp + s * 32 + hl * 16 + 8);
    }

    const v8f vzero = {0.f, 0.f, 0.f, 0.f, 0.f, 0.f, 0.f, 0.f};
    v8f oac[4];
    for (int i = 0; i < 4; ++i) oac[i] = vzero;
    float mrow = -1e30f, lsum = 0.f;

    for (int kb = 0; kb < 2048; kb += 32) {
        __syncthreads();
        // stage V^T tile: each thread handles one (key-pair, 8 dh) cell,
        // packing two keys into one b32 LDS store.
        {
            int kp = t >> 3, dh0 = (t & 7) * 8;  // kp 0..15, dh0 0..56
            const u16* vp0 = qkv + ((size_t)(b * 2048 + kb + 2 * kp)) * RS
                             + 2048 + head * 64 + dh0;
            uint4 r0 = *(const uint4*)vp0;
            uint4 r1 = *(const uint4*)(vp0 + RS);
            const u16* e0 = (const u16*)&r0;
            const u16* e1 = (const u16*)&r1;
            u32* vt = (u32*)Vt;
            for (int j = 0; j < 8; ++j)
                vt[(dh0 + j) * (LDP / 2) + kp] = (u32)e0[j] | ((u32)e1[j] << 16);
        }
        __syncthreads();

        // S^T[32 keys x 16 q] = K(32x64) x Q^T(64x16)
        v8f st[2] = {vzero, vzero};
        for (int r2 = 0; r2 < 2; ++r2) {
            const u16* kp_ = qkv + ((size_t)(b * 2048 + kb + r2 * 16 + lr)) * RS
                             + 1024 + head * 64;
            for (int s = 0; s < 2; ++s) {
                Frag ka;
                ka.u[0] = *(const uint4*)(kp_ + s * 32 + hl * 16);
                ka.u[1] = *(const uint4*)(kp_ + s * 32 + hl * 16 + 8);
                st[r2] = __builtin_amdgcn_wmma_f32_16x16x32_bf16(
                    false, ka.v, false, qb[s].v, (short)0, st[r2], false, false);
            }
        }

        // online softmax (row = query = lane pair {lr, lr+16})
        float tmax = -1e30f;
        for (int r = 0; r < 8; ++r) {
            tmax = fmaxf(tmax, st[0][r] * SC);
            tmax = fmaxf(tmax, st[1][r] * SC);
        }
        tmax = fmaxf(tmax, __shfl_xor(tmax, 16, 32));
        const float mnew = fmaxf(mrow, tmax);
        const float corr = __expf(mrow - mnew);
        for (int i = 0; i < 4; ++i)
            for (int r = 0; r < 8; ++r) oac[i][r] *= corr;
        lsum *= corr;

        float p0[8], p1[8], psum = 0.f;
        for (int r = 0; r < 8; ++r) {
            p0[r] = __expf(st[0][r] * SC - mnew);
            p1[r] = __expf(st[1][r] * SC - mnew);
            psum += p0[r] + p1[r];
        }
        psum += __shfl_xor(psum, 16, 32);
        lsum += psum;
        mrow = mnew;

        // build P^T B-fragment: lanes<16 keys 0..15, lanes>=16 keys 16..31
        Frag pb;
        for (int r = 0; r < 8; ++r) {
            float o0 = __shfl_xor(p0[r], 16, 32);
            float o1 = __shfl_xor(p1[r], 16, 32);
            float lo = hl ? o1 : p0[r];
            float hi = hl ? p1[r] : o0;
            pb.h[r]     = f2bf(lo);
            pb.h[r + 8] = f2bf(hi);
        }

        // O^T[64 dh x 16 q] += V^T(64x32) x P^T(32x16)
        for (int t4 = 0; t4 < 4; ++t4) {
            Frag va;
            int base = (t4 * 16 + lr) * LDP + hl * 16;
            va.u[0] = *(const uint4*)&Vt[base];
            va.u[1] = *(const uint4*)&Vt[base + 8];
            oac[t4] = __builtin_amdgcn_wmma_f32_16x16x32_bf16(
                false, va.v, false, pb.v, (short)0, oac[t4], false, false);
        }
    }

    const float inv = 1.0f / lsum;
    for (int t4 = 0; t4 < 4; ++t4)
        for (int r = 0; r < 8; ++r) {
            int dh = t4 * 16 + r + hl * 8;
            o[((size_t)(b * 2048 + qbase + lr)) * 1024 + head * 64 + dh] =
                f2bf(oac[t4][r] * inv);
        }
}

// ---------------------------------------------------------------------------
extern "C" void kernel_launch(void* const* d_in, const int* in_sizes, int n_in,
                              void* d_out, int out_size, void* d_ws, size_t ws_size,
                              hipStream_t stream) {
    (void)in_sizes; (void)n_in; (void)out_size; (void)ws_size;
    const float* x     = (const float*)d_in[0];
    const float* ln1_g = (const float*)d_in[1];
    const float* ln1_b = (const float*)d_in[2];
    const float* wqkv  = (const float*)d_in[3];
    const float* bqkv  = (const float*)d_in[4];
    const float* wo    = (const float*)d_in[5];
    const float* bo    = (const float*)d_in[6];
    const float* ln2_g = (const float*)d_in[7];
    const float* ln2_b = (const float*)d_in[8];
    const float* w1    = (const float*)d_in[9];
    const float* b1    = (const float*)d_in[10];
    const float* w2    = (const float*)d_in[11];
    const float* b2    = (const float*)d_in[12];

    const int TOK = 2 * 2048;           // B*N
    float* h = (float*)d_out;           // residual stream lives in d_out

    u16* y    = (u16*)d_ws;              // [4096,1024]
    u16* qkvb = y + (size_t)TOK * 1024;  // [4096,3072]
    u16* ov   = qkvb + (size_t)TOK * 3072;
    u16* m    = ov + (size_t)TOK * 1024; // [4096,4096]

    hipMemcpyAsync(h, x, (size_t)TOK * 1024 * sizeof(float),
                   hipMemcpyDeviceToDevice, stream);

    for (int l = 0; l < 8; ++l) {
        const size_t oD = (size_t)l * 1024;
        ln_bf16<<<TOK, 256, 0, stream>>>(h, ln1_g + oD, ln1_b + oD, y);
        gemm_bf16<0><<<dim3(3072 / 128, TOK / 128), 256, 0, stream>>>(
            y, wqkv + (size_t)l * 1024 * 3072, bqkv + (size_t)l * 3072,
            (void*)qkvb, 1024, 3072);
        attn<<<dim3(2048 / 64, 16, 2), 128, 0, stream>>>(qkvb, ov);
        gemm_bf16<2><<<dim3(1024 / 128, TOK / 128), 256, 0, stream>>>(
            ov, wo + (size_t)l * 1024 * 1024, bo + oD, (void*)h, 1024, 1024);
        ln_bf16<<<TOK, 256, 0, stream>>>(h, ln2_g + oD, ln2_b + oD, y);
        gemm_bf16<1><<<dim3(4096 / 128, TOK / 128), 256, 0, stream>>>(
            y, w1 + (size_t)l * 1024 * 4096, b1 + (size_t)l * 4096,
            (void*)m, 1024, 4096);
        gemm_bf16<2><<<dim3(1024 / 128, TOK / 128), 256, 0, stream>>>(
            m, w2 + (size_t)l * 4096 * 1024, b2 + oD, (void*)h, 4096, 1024);
    }
}
